// MultiHeadAttention_3968549782214
// MI455X (gfx1250) — matrix operand order repaired and checked by reading, not compiled, not run on hardware
//
#include <hip/hip_runtime.h>

// Problem dims
#define S_  4096
#define D_  2048
#define H_  16
#define DK_ 128
#define DV_ 128
#define HDV_ (H_ * DV_)   // 2048

typedef __attribute__((ext_vector_type(16))) _Float16 v16h;
typedef __attribute__((ext_vector_type(8)))  _Float16 v8h;
typedef __attribute__((ext_vector_type(8)))  float    v8f;

typedef __attribute__((ext_vector_type(4))) unsigned int tdm_g0_t;
typedef __attribute__((ext_vector_type(8))) int          tdm_g1_t;
typedef __attribute__((ext_vector_type(4))) int          tdm_g2_t;

static __device__ __forceinline__ v8h ld8(const _Float16* p) {
  return *reinterpret_cast<const v8h*>(p);
}
static __device__ __forceinline__ v16h mk16(v8h lo, v8h hi) {
  v16h r;
#pragma unroll
  for (int i = 0; i < 8; ++i) { r[i] = lo[i]; r[i + 8] = hi[i]; }
  return r;
}
static __device__ __forceinline__ v8f wmma16(v16h a, v16h b, v8f c) {
  // D = A(16x32 f16) * B(32x16 f16) + C(16x16 f32)
  return __builtin_amdgcn_wmma_f32_16x16x32_f16(
      false, a, false, b, (short)0, c, false, false);
}

// ---------------------------------------------------------------------------
// TDM: issue a 2-D (or 1-D when tile_d1==0) f16 tile load Global -> LDS.
// D# bitfields per CDNA5 ISA ch.8: group0 = {count/flags, lds_addr, gaddr,
// gaddr_hi|type=2}; group1 packs data_size=2B, tensor dims, tile dims, stride.
// ---------------------------------------------------------------------------
static __device__ __forceinline__ void tdm_load_f16(
    unsigned lds_addr, const _Float16* gptr,
    unsigned tensor_d0, unsigned tensor_d1,
    unsigned tile_d0,   unsigned tile_d1,
    unsigned long long row_stride_elems)
{
  unsigned long long ga = (unsigned long long)(uintptr_t)gptr;
  tdm_g0_t g0;
  g0[0] = 1u;                                            // count=1, user desc
  g0[1] = lds_addr;                                      // LDS byte address
  g0[2] = (unsigned)(ga & 0xffffffffu);                  // global_addr[31:0]
  g0[3] = (unsigned)((ga >> 32) & 0x1ffffffu) | (2u << 30);  // [56:32] | type=2
  tdm_g1_t g1;
  g1[0] = (int)0x00010000u;                              // data_size=1 (2 bytes)
  g1[1] = (int)((tensor_d0 & 0xffffu) << 16);            // abar=0 | tdim0 lo16
  g1[2] = (int)(((tensor_d0 >> 16) & 0xffffu) | ((tensor_d1 & 0xffffu) << 16));
  g1[3] = (int)(((tensor_d1 >> 16) & 0xffffu) | ((tile_d0 & 0xffffu) << 16));
  g1[4] = (int)(tile_d1 & 0xffffu);                      // tile_dim1 | tile_dim2=0
  g1[5] = (int)(row_stride_elems & 0xffffffffull);       // dim0_stride[31:0]
  g1[6] = (int)((row_stride_elems >> 32) & 0xffffull);   // dim0_stride[47:32]
  g1[7] = 0;                                             // dim1_stride=0 (2D)
  tdm_g2_t z4 = {0, 0, 0, 0};
#if defined(__clang_major__) && __clang_major__ >= 23
  tdm_g1_t z8 = {0, 0, 0, 0, 0, 0, 0, 0};
  __builtin_amdgcn_tensor_load_to_lds(g0, g1, z4, z4, z8, 0);
#else
  __builtin_amdgcn_tensor_load_to_lds(g0, g1, z4, z4, 0);
#endif
}

// ---------------------------------------------------------------------------
// Prep kernels: f32 -> f16 conversions + weight transposes (column-major B)
// ---------------------------------------------------------------------------
__global__ void cvt_kernel(const float* __restrict__ in, _Float16* __restrict__ out, int n) {
  for (int i = blockIdx.x * blockDim.x + threadIdx.x; i < n; i += gridDim.x * blockDim.x)
    out[i] = (_Float16)in[i];
}

// [H, D, 128] f32 -> [H, 128, D] f16  (column-major per head)
__global__ void transpose_heads_kernel(const float* __restrict__ W, _Float16* __restrict__ WT) {
  const int total = H_ * D_ * DK_;
  for (int i = blockIdx.x * blockDim.x + threadIdx.x; i < total; i += gridDim.x * blockDim.x) {
    int h = i / (D_ * DK_);
    int r = i % (D_ * DK_);
    int d = r / DK_;
    int n = r % DK_;
    WT[(size_t)h * DK_ * D_ + (size_t)n * D_ + d] = (_Float16)W[i];
  }
}

// Wo [HDV, D] f32 -> WoT [D, HDV] f16
__global__ void transpose_wo_kernel(const float* __restrict__ Wo, _Float16* __restrict__ WoT) {
  const int total = HDV_ * D_;
  for (int i = blockIdx.x * blockDim.x + threadIdx.x; i < total; i += gridDim.x * blockDim.x) {
    int k = i / D_;
    int n = i % D_;
    WoT[(size_t)n * HDV_ + k] = (_Float16)Wo[i];
  }
}

// ---------------------------------------------------------------------------
// Fused QKV projection GEMM: one 16x64 output strip per wave (4 N-tiles, A
// fragment reused across 4 WMMAs). K = D_ = 2048.
// ---------------------------------------------------------------------------
__global__ __launch_bounds__(256) void proj_kernel(
    const _Float16* __restrict__ xh,                 // [S, D]
    const _Float16* __restrict__ WqT,                // [H, DK, D]
    const _Float16* __restrict__ WkT,
    const _Float16* __restrict__ WvT,
    const float* __restrict__ bq, const float* __restrict__ bk, const float* __restrict__ bv,
    _Float16* __restrict__ Qh,                       // [H, S, DK]
    _Float16* __restrict__ Kh,                       // [H, S, DK]
    _Float16* __restrict__ VhT)                      // [H, DV, S]
{
  const int wave = blockIdx.x * (blockDim.x >> 5) + (threadIdx.x >> 5);
  const int lane = threadIdx.x & 31;
  const int col  = lane & 15;
  const int hi   = lane >> 4;

  const int NG  = DK_ / 64;                          // 2 N-groups of 64
  const int TPM = H_ * (S_ / 16) * NG;               // 8192 strips per matrix
  const int mat = wave / TPM;                        // 0=Q 1=K 2=V
  int r0 = wave % TPM;
  const int h  = r0 / ((S_ / 16) * NG);
  int r1 = r0 % ((S_ / 16) * NG);
  const int m0 = (r1 / NG) * 16;
  const int n0 = (r1 % NG) * 64;

  const _Float16* WT   = (mat == 0) ? WqT : (mat == 1) ? WkT : WvT;
  const float*    bias = (mat == 0) ? bq  : (mat == 1) ? bk  : bv;

  const _Float16* aptr = xh + (size_t)(m0 + col) * D_;
  const _Float16* bptr[4];
#pragma unroll
  for (int j = 0; j < 4; ++j)
    bptr[j] = WT + ((size_t)h * DK_ + n0 + j * 16 + col) * D_;

  const int a0 = hi * 8, a1 = 16 + hi * 8;           // A-frag K halves
  const int b0 = hi * 8;                             // B-frag K halves

  v8f acc[4];
#pragma unroll
  for (int j = 0; j < 4; ++j) acc[j] = (v8f){};

  for (int k0 = 0; k0 < D_; k0 += 32) {
    __builtin_prefetch(aptr + k0 + 256, 0, 0);
    v16h a = mk16(ld8(aptr + k0 + a0), ld8(aptr + k0 + a1));
#pragma unroll
    for (int j = 0; j < 4; ++j) {
      __builtin_prefetch(bptr[j] + k0 + 256, 0, 0);
      v16h b = mk16(ld8(bptr[j] + k0 + b0), ld8(bptr[j] + k0 + b0 + 16));
      acc[j] = wmma16(a, b, acc[j]);
    }
  }

#pragma unroll
  for (int j = 0; j < 4; ++j) {
    const int n = n0 + j * 16 + col;
    const float bn = bias[h * DK_ + n];
    if (mat < 2) {
      _Float16* outp = (mat == 0 ? Qh : Kh) + (size_t)h * S_ * DK_;
#pragma unroll
      for (int r = 0; r < 8; ++r) {
        int m = m0 + r + hi * 8;
        outp[(size_t)m * DK_ + n] = (_Float16)(acc[j][r] + bn);
      }
    } else {
#pragma unroll
      for (int r = 0; r < 8; ++r) {
        int m = m0 + r + hi * 8;
        VhT[((size_t)h * DV_ + n) * S_ + m] = (_Float16)(acc[j][r] + bn);
      }
    }
  }
}

// ---------------------------------------------------------------------------
// Flash attention: one wave per (16-query tile, head). K/V chunks staged into
// LDS by the Tensor Data Mover, double-buffered on TENSORcnt. Online softmax;
// P bounced through LDS (C-layout -> A-layout). Output written straight into
// the concatenated [S, H*DV] f16 activation.
// ---------------------------------------------------------------------------
__global__ __launch_bounds__(32) void attn_kernel(
    const _Float16* __restrict__ Qh,   // [H, S, DK]
    const _Float16* __restrict__ Kh,   // [H, S, DK]
    const _Float16* __restrict__ VhT,  // [H, DV, S]
    _Float16* __restrict__ catC)       // [S, H*DV]
{
  __shared__ __align__(16) _Float16 kLds[2][32 * DK_];  // 2 x 8 KB
  __shared__ __align__(16) _Float16 vLds[2][DV_ * 32];  // 2 x 8 KB
  __shared__ __align__(16) _Float16 pLds[16 * 32];      // 1 KB

  const int lane = threadIdx.x & 31;
  const int col  = lane & 15;
  const int hi   = lane >> 4;
  const int q0   = blockIdx.x * 16;
  const int h    = blockIdx.y;

  const _Float16* kbase = Kh  + (size_t)h * S_ * DK_;
  const _Float16* vbase = VhT + (size_t)h * DV_ * S_;

  // Preload the 16x128 Q tile as 4 A-fragments (resident for whole kernel)
  const _Float16* qptr = Qh + ((size_t)h * S_ + q0 + col) * DK_;
  v16h qf[4];
#pragma unroll
  for (int kk = 0; kk < 4; ++kk)
    qf[kk] = mk16(ld8(qptr + kk * 32 + hi * 8), ld8(qptr + kk * 32 + 16 + hi * 8));

  v8f zero = {};
  v8f acc[8];
#pragma unroll
  for (int t = 0; t < 8; ++t) acc[t] = zero;
  float mrun[8], lrun[8];
#pragma unroll
  for (int r = 0; r < 8; ++r) { mrun[r] = -1e30f; lrun[r] = 0.0f; }

  const float scale = 0.088388347648318447f;  // 1/sqrt(128)
  const int NC = S_ / 32;                     // 128 key chunks

  // Issue TDM loads for key-chunk c into buffer c&1:
  //  K chunk: rows [s0, s0+32) of Kh[h], contiguous -> 1-D tile of 4096 f16.
  //  V chunk: cols [s0, s0+32) of VhT[h] (128 rows, stride S_) -> 2-D tile.
  auto issue = [&](int c) {
    const int s0 = c * 32;
    const int buf = c & 1;
    tdm_load_f16((unsigned)(uintptr_t)&kLds[buf][0], kbase + (size_t)s0 * DK_,
                 /*tensor_d0=*/32 * DK_, /*tensor_d1=*/1,
                 /*tile_d0=*/32 * DK_,   /*tile_d1=*/0,
                 /*row_stride=*/32 * DK_);
    tdm_load_f16((unsigned)(uintptr_t)&vLds[buf][0], vbase + s0,
                 /*tensor_d0=*/32,  /*tensor_d1=*/DV_,
                 /*tile_d0=*/32,    /*tile_d1=*/DV_,
                 /*row_stride=*/(unsigned long long)S_);
  };

  auto compute = [&](int c) {
    const int buf = c & 1;
    const _Float16* kc = &kLds[buf][0];
    const _Float16* vc = &vLds[buf][0];

    // --- scores: Sc[16q x 32keys] = Q @ K^T (two 16x16 C tiles, 4 k-steps) ---
    v8f sc[2]; sc[0] = zero; sc[1] = zero;
#pragma unroll
    for (int t = 0; t < 2; ++t) {
      const _Float16* kp = kc + (size_t)(t * 16 + col) * DK_;
#pragma unroll
      for (int kk = 0; kk < 4; ++kk) {
        v16h b = mk16(ld8(kp + kk * 32 + hi * 8), ld8(kp + kk * 32 + 16 + hi * 8));
        sc[t] = wmma16(qf[kk], b, sc[t]);
      }
    }

    // --- online softmax: per-row (half-wave) reductions, wave32-safe ---
    float p0[8], p1[8], alpha[8];
#pragma unroll
    for (int r = 0; r < 8; ++r) {
      float a = sc[0][r] * scale, b = sc[1][r] * scale;
      float cm = fmaxf(a, b);
      cm = fmaxf(cm, __shfl_xor(cm, 1, 32));
      cm = fmaxf(cm, __shfl_xor(cm, 2, 32));
      cm = fmaxf(cm, __shfl_xor(cm, 4, 32));
      cm = fmaxf(cm, __shfl_xor(cm, 8, 32));
      float mnew = fmaxf(mrun[r], cm);
      alpha[r] = __expf(mrun[r] - mnew);
      p0[r] = __expf(a - mnew);
      p1[r] = __expf(b - mnew);
      float rs = p0[r] + p1[r];
      rs += __shfl_xor(rs, 1, 32);
      rs += __shfl_xor(rs, 2, 32);
      rs += __shfl_xor(rs, 4, 32);
      rs += __shfl_xor(rs, 8, 32);
      lrun[r] = lrun[r] * alpha[r] + rs;
      mrun[r] = mnew;
    }
#pragma unroll
    for (int t = 0; t < 8; ++t)
#pragma unroll
      for (int r = 0; r < 8; ++r) acc[t][r] *= alpha[r];

    // --- C-layout P -> LDS -> A-layout fragment ---
#pragma unroll
    for (int r = 0; r < 8; ++r) {
      pLds[(r + hi * 8) * 32 + col]      = (_Float16)p0[r];
      pLds[(r + hi * 8) * 32 + 16 + col] = (_Float16)p1[r];
    }
    __syncthreads();
    v16h af = mk16(ld8(&pLds[col * 32 + hi * 8]), ld8(&pLds[col * 32 + 16 + hi * 8]));

    // --- PV: acc[16q x 128v] += P(16x32) @ V(32x128), 8 B tiles from LDS ---
#pragma unroll
    for (int t = 0; t < 8; ++t) {
      const _Float16* vp = vc + (size_t)(t * 16 + col) * 32 + hi * 8;
      v16h b = mk16(ld8(vp), ld8(vp + 16));
      acc[t] = wmma16(af, b, acc[t]);
    }
    __syncthreads();
  };

  issue(0);
  issue(1);
  for (int c = 0; c < NC; ++c) {
    if (c + 1 < NC) __builtin_amdgcn_s_wait_tensorcnt(2);  // chunk c landed
    else            __builtin_amdgcn_s_wait_tensorcnt(0);  // last chunk
    asm volatile("" ::: "memory");
    compute(c);
    asm volatile("" ::: "memory");
    if (c + 2 < NC) issue(c + 2);
  }

  // --- epilogue: normalize by l, write into concatenated activation ---
#pragma unroll
  for (int r = 0; r < 8; ++r) {
    float inv = 1.0f / lrun[r];
    int srow = q0 + r + hi * 8;
#pragma unroll
    for (int t = 0; t < 8; ++t)
      catC[(size_t)srow * HDV_ + h * DV_ + t * 16 + col] = (_Float16)(acc[t][r] * inv);
  }
}

// ---------------------------------------------------------------------------
// Output projection: out[S, D] f32 = catC[S, 2048] f16 @ Wo + bo
// One 16x64 strip per wave (A fragment reused across 4 WMMAs).
// ---------------------------------------------------------------------------
__global__ __launch_bounds__(256) void outproj_kernel(
    const _Float16* __restrict__ catC,  // [S, HDV]
    const _Float16* __restrict__ WoT,   // [D, HDV] (column-major Wo)
    const float* __restrict__ bo,
    float* __restrict__ out)            // [S, D]
{
  const int wave = blockIdx.x * (blockDim.x >> 5) + (threadIdx.x >> 5);
  const int lane = threadIdx.x & 31;
  const int col  = lane & 15;
  const int hi   = lane >> 4;

  const int NG = D_ / 64;             // 32 strips along N
  const int m0 = (wave / NG) * 16;
  const int n0 = (wave % NG) * 64;

  const _Float16* aptr = catC + (size_t)(m0 + col) * HDV_;
  const _Float16* bptr[4];
#pragma unroll
  for (int j = 0; j < 4; ++j)
    bptr[j] = WoT + (size_t)(n0 + j * 16 + col) * HDV_;

  v8f acc[4];
#pragma unroll
  for (int j = 0; j < 4; ++j) acc[j] = (v8f){};

  for (int k0 = 0; k0 < HDV_; k0 += 32) {
    __builtin_prefetch(aptr + k0 + 256, 0, 0);
    v16h a = mk16(ld8(aptr + k0 + hi * 8), ld8(aptr + k0 + 16 + hi * 8));
#pragma unroll
    for (int j = 0; j < 4; ++j) {
      __builtin_prefetch(bptr[j] + k0 + 256, 0, 0);
      v16h b = mk16(ld8(bptr[j] + k0 + hi * 8), ld8(bptr[j] + k0 + 16 + hi * 8));
      acc[j] = wmma16(a, b, acc[j]);
    }
  }

#pragma unroll
  for (int j = 0; j < 4; ++j) {
    const float bn = bo[n0 + j * 16 + col];
#pragma unroll
    for (int r = 0; r < 8; ++r)
      out[(size_t)(m0 + r + hi * 8) * D_ + n0 + j * 16 + col] = acc[j][r] + bn;
  }
}

// ---------------------------------------------------------------------------
extern "C" void kernel_launch(void* const* d_in, const int* in_sizes, int n_in,
                              void* d_out, int out_size, void* d_ws, size_t ws_size,
                              hipStream_t stream) {
  (void)in_sizes; (void)n_in; (void)out_size; (void)ws_size;

  const float* x  = (const float*)d_in[0];
  const float* Wq = (const float*)d_in[1];
  const float* bq = (const float*)d_in[2];
  const float* Wk = (const float*)d_in[3];
  const float* bk = (const float*)d_in[4];
  const float* Wv = (const float*)d_in[5];
  const float* bv = (const float*)d_in[6];
  const float* Wo = (const float*)d_in[7];
  const float* bo = (const float*)d_in[8];
  float* out = (float*)d_out;

  char* ws = (char*)d_ws;
  size_t off = 0;
  _Float16* xh   = (_Float16*)(ws + off); off += (size_t)S_ * D_ * 2;        // 16 MB
  _Float16* WqT  = (_Float16*)(ws + off); off += (size_t)H_ * DK_ * D_ * 2;  //  8 MB
  _Float16* WkT  = (_Float16*)(ws + off); off += (size_t)H_ * DK_ * D_ * 2;  //  8 MB
  _Float16* WvT  = (_Float16*)(ws + off); off += (size_t)H_ * DV_ * D_ * 2;  //  8 MB
  _Float16* WoT  = (_Float16*)(ws + off); off += (size_t)D_ * HDV_ * 2;      //  8 MB
  _Float16* Qh   = (_Float16*)(ws + off); off += (size_t)H_ * S_ * DK_ * 2;  // 16 MB
  _Float16* Kh   = (_Float16*)(ws + off); off += (size_t)H_ * S_ * DK_ * 2;  // 16 MB
  _Float16* VhT  = (_Float16*)(ws + off); off += (size_t)H_ * DV_ * S_ * 2;  // 16 MB
  _Float16* catC = (_Float16*)(ws + off); off += (size_t)S_ * HDV_ * 2;      // 16 MB

  // Stage 0: precision conversion + B-operand transposes
  cvt_kernel<<<4096, 256, 0, stream>>>(x, xh, S_ * D_);
  transpose_heads_kernel<<<4096, 256, 0, stream>>>(Wq, WqT);
  transpose_heads_kernel<<<4096, 256, 0, stream>>>(Wk, WkT);
  transpose_heads_kernel<<<4096, 256, 0, stream>>>(Wv, WvT);
  transpose_wo_kernel<<<4096, 256, 0, stream>>>(Wo, WoT);

  // Stage 1: fused QKV projection (3 * 8192 strips, 8 waves/block)
  proj_kernel<<<(3 * H_ * (S_ / 16) * (DK_ / 64)) / 8, 256, 0, stream>>>(
      xh, WqT, WkT, WvT, bq, bk, bv, Qh, Kh, VhT);

  // Stage 2: flash attention, one wave per (q-tile, head), TDM-staged K/V
  attn_kernel<<<dim3(S_ / 16, H_), 32, 0, stream>>>(Qh, Kh, VhT, catC);

  // Stage 3: output projection (8192 strips, 8 waves/block)
  outproj_kernel<<<((S_ / 16) * (D_ / 64)) / 8, 256, 0, stream>>>(catC, WoT, bo, out);
}
